// XlaQuantizedLinear_42966852829503
// MI455X (gfx1250) — compile-verified, matrix operand order checked by reading, not run on hardware
//
#include <hip/hip_runtime.h>

typedef _Float16 v4h  __attribute__((ext_vector_type(4)));
typedef _Float16 v8h  __attribute__((ext_vector_type(8)));
typedef _Float16 v16h __attribute__((ext_vector_type(16)));
typedef float    v8f  __attribute__((ext_vector_type(8)));

#define D_IN   4096
#define D_OUT  11008
#define M_TOT  8192
#define NBLK   32      // K blocks of 128
#define BM     128
#define BN     128
#define KS     64      // K elements staged per iteration (2 WMMA k-steps)
#define XST    72      // LDS row stride in halfs for xs/ws (padded, conflict-free)
#define ZST    40      // LDS row stride for zpT
#define SST    33      // LDS row stride (floats) for xsum

__global__ __launch_bounds__(256)
void qmm_wmma_kernel(const float* __restrict__ x,
                     const int*   __restrict__ w,
                     const float* __restrict__ scaler,
                     const float* __restrict__ zp,
                     float*       __restrict__ out)
{
    __shared__ _Float16 xs[BM * XST];      // x tile, fp16, row-major [m][k]
    __shared__ _Float16 ws[BN * XST];      // scaled w tile, fp16, [n][k] (transposed)
    __shared__ float    xsum[BM * SST];    // per-row, per-128-block sums of x
    __shared__ _Float16 zpT[BN * ZST];     // -zero_point, fp16, [n][s]

    const int tid  = threadIdx.x;
    const int lane = tid & 31;
    const int wv   = tid >> 5;      // wave 0..7
    const int wm   = wv >> 1;       // 0..3 : 32-row slab
    const int wn   = wv & 1;        // 0..1 : 64-col slab
    const int lh   = lane & 15;
    const int hi   = lane >> 4;     // lane group 0/1 (K-offset +8 per ISA layout)

    const int n0 = blockIdx.x * BN;
    const int m0 = blockIdx.y * BM;

    // ---- one-time: stage -zero_point (fp16, transposed) and zero xsum
    for (int i = 0; i < (NBLK * BN) / 256; ++i) {
        int e = tid + i * 256;
        int s = e >> 7;            // /128
        int n = e & 127;
        zpT[n * ZST + s] = (_Float16)(-zp[(size_t)s * D_OUT + n0 + n]);
    }
    for (int idx = tid; idx < BM * SST; idx += 256) xsum[idx] = 0.0f;

    // accumulators: 2 m-tiles x 4 n-tiles of 16x16 f32
    v8f acc[2][4];
    {
        v8f z = {0.f,0.f,0.f,0.f,0.f,0.f,0.f,0.f};
        for (int a = 0; a < 2; ++a)
            for (int b = 0; b < 4; ++b)
                acc[a][b] = z;
    }

    // per-thread fixed load coordinates (coalesced 256B segments)
    const int xrow = tid >> 4;            // 0..15  (+16*i)
    const int xcol = (tid & 15) * 4;      // 0..60
    const int wrow = tid >> 5;            // 0..7   (+8*i)
    const int wcol = (tid & 31) * 4;      // 0..124

    float4 xr[8];
    int4   wr[8];
    float4 scl;

    // prefetch stage 0 into registers
    {
        const float* xp = x + (size_t)(m0 + xrow) * D_IN + xcol;
        for (int i = 0; i < 8; ++i)
            xr[i] = *(const float4*)(xp + (size_t)i * 16 * D_IN);
        const int* wp = w + (size_t)wrow * D_OUT + n0 + wcol;
        for (int i = 0; i < 8; ++i)
            wr[i] = *(const int4*)(wp + (size_t)i * 8 * D_OUT);
        scl = *(const float4*)(scaler + (size_t)0 * D_OUT + n0 + wcol);
    }

    for (int kk = 0; kk < D_IN; kk += KS) {
        __syncthreads();   // previous stage fully consumed

        // regs -> LDS: x fp32->fp16 ; w int32 * scaler -> fp16, transposed
        for (int i = 0; i < 8; ++i) {
            v4h h;
            h[0] = (_Float16)xr[i].x; h[1] = (_Float16)xr[i].y;
            h[2] = (_Float16)xr[i].z; h[3] = (_Float16)xr[i].w;
            *(v4h*)&xs[(xrow + i * 16) * XST + xcol] = h;
        }
        for (int i = 0; i < 8; ++i) {
            const int kr = wrow + i * 8;
            ws[(wcol + 0) * XST + kr] = (_Float16)((float)wr[i].x * scl.x);
            ws[(wcol + 1) * XST + kr] = (_Float16)((float)wr[i].y * scl.y);
            ws[(wcol + 2) * XST + kr] = (_Float16)((float)wr[i].z * scl.z);
            ws[(wcol + 3) * XST + kr] = (_Float16)((float)wr[i].w * scl.w);
        }

        __syncthreads();   // LDS ready

        // prefetch next stage (overlaps HBM latency with WMMAs below)
        const int kn = kk + KS;
        if (kn < D_IN) {
            const float* xp = x + (size_t)(m0 + xrow) * D_IN + kn + xcol;
            for (int i = 0; i < 8; ++i)
                xr[i] = *(const float4*)(xp + (size_t)i * 16 * D_IN);
            const int* wp = w + (size_t)(kn + wrow) * D_OUT + n0 + wcol;
            for (int i = 0; i < 8; ++i)
                wr[i] = *(const int4*)(wp + (size_t)i * 8 * D_OUT);
            scl = *(const float4*)(scaler + (size_t)(kn >> 7) * D_OUT + n0 + wcol);
        }

        // accumulate per-row sums for the zero-point term (stage is within one 128-block)
        if (tid < BM) {
            float s = 0.0f;
            for (int j = 0; j < 8; ++j) {
                v8h v = *(const v8h*)&xs[tid * XST + j * 8];
                for (int e = 0; e < 8; ++e) s += (float)v[e];
            }
            xsum[tid * SST + (kk >> 7)] += s;
        }

        // two K=32 WMMA steps over the staged 64-wide slab
        for (int ks = 0; ks < 2; ++ks) {
            const int ko = ks * 32 + hi * 8;
            union { v16h v; v8h h[2]; } A[2], B[4];
            for (int mt = 0; mt < 2; ++mt) {
                const int r = wm * 32 + mt * 16 + lh;
                A[mt].h[0] = *(const v8h*)&xs[r * XST + ko];
                A[mt].h[1] = *(const v8h*)&xs[r * XST + ko + 16];
            }
            for (int nt = 0; nt < 4; ++nt) {
                const int c = wn * 64 + nt * 16 + lh;
                B[nt].h[0] = *(const v8h*)&ws[c * XST + ko];
                B[nt].h[1] = *(const v8h*)&ws[c * XST + ko + 16];
            }
            for (int mt = 0; mt < 2; ++mt)
                for (int nt = 0; nt < 4; ++nt)
                    acc[mt][nt] = __builtin_amdgcn_wmma_f32_16x16x32_f16(
                        false, A[mt].v, false, B[nt].v,
                        (short)0, acc[mt][nt], false, false);
        }
    }

    __syncthreads();   // xsum fully accumulated

    // ---- zero-point correction: one extra K=32 WMMA per C tile
    {
        const int sb = hi * 8;
        union { v16h v; _Float16 e[16]; } A[2];
        for (int mt = 0; mt < 2; ++mt) {
            const int r = wm * 32 + mt * 16 + lh;
            for (int j = 0; j < 8; ++j) {
                A[mt].e[j]     = (_Float16)xsum[r * SST + sb + j];
                A[mt].e[8 + j] = (_Float16)xsum[r * SST + 16 + sb + j];
            }
        }
        union { v16h v; v8h h[2]; } B[4];
        for (int nt = 0; nt < 4; ++nt) {
            const int c = wn * 64 + nt * 16 + lh;
            B[nt].h[0] = *(const v8h*)&zpT[c * ZST + sb];
            B[nt].h[1] = *(const v8h*)&zpT[c * ZST + sb + 16];
        }
        for (int mt = 0; mt < 2; ++mt)
            for (int nt = 0; nt < 4; ++nt)
                acc[mt][nt] = __builtin_amdgcn_wmma_f32_16x16x32_f16(
                    false, A[mt].v, false, B[nt].v,
                    (short)0, acc[mt][nt], false, false);
    }

    // ---- store C tiles (VGPR r -> row r, lanes 16-31 -> rows +8)
    for (int mt = 0; mt < 2; ++mt) {
        const int mb = m0 + wm * 32 + mt * 16 + hi * 8;
        for (int nt = 0; nt < 4; ++nt) {
            const int n = n0 + wn * 64 + nt * 16 + lh;
            for (int r = 0; r < 8; ++r)
                out[(size_t)(mb + r) * D_OUT + n] = acc[mt][nt][r];
        }
    }
}

extern "C" void kernel_launch(void* const* d_in, const int* in_sizes, int n_in,
                              void* d_out, int out_size, void* d_ws, size_t ws_size,
                              hipStream_t stream) {
    (void)in_sizes; (void)n_in; (void)out_size; (void)d_ws; (void)ws_size;
    const float* x      = (const float*)d_in[0];
    const int*   wgt    = (const int*)d_in[1];
    const float* scaler = (const float*)d_in[2];
    const float* zpnt   = (const float*)d_in[3];
    float* out = (float*)d_out;

    dim3 grid(D_OUT / BN, M_TOT / BM);   // 86 x 64 workgroups
    dim3 block(256);                      // 8 wave32 waves
    hipLaunchKernelGGL(qmm_wmma_kernel, grid, block, 0, stream,
                       x, wgt, scaler, zpnt, out);
}
